// MultiHeadDotProductAttention_19129784336436
// MI455X (gfx1250) — matrix operand order repaired and checked by reading, not compiled, not run on hardware
//
#include <hip/hip_runtime.h>

// ---------------------------------------------------------------------------
// Problem constants
// ---------------------------------------------------------------------------
#define BB   1024
#define SL   79          // sequence length
#define EE   512         // embedding
#define HH   8           // heads
#define DH   64          // head dim
#define SUMD 128         // smolgen summary dim
#define SS   (SL * SL)   // 6241
#define SSPAD 6272       // 49 * 128
#define MROWS (BB * SL)  // 80896 = 1264 * 64

typedef __attribute__((ext_vector_type(16))) _Float16 v16h;
typedef __attribute__((ext_vector_type(8)))  _Float16 v8h;
typedef __attribute__((ext_vector_type(8)))  float    v8f;

// ---------------------------------------------------------------------------
// LDS-tiled WMMA GEMM:  C[M,N] = A[M,K] * B[K,N], B supplied transposed
// (BT[Npad,K] row-major).  Block = 256 threads = 8 waves arranged 4(M) x 2(N);
// block tile 64x128, wave tile 16x64 (4 accumulators).  Per 32-K chunk the
// block stages A(64x32) + BT(128x32) panels into LDS with
// global_load_async_to_lds_b128 (ASYNCcnt), double buffered; fragments are
// then read from LDS (ds_load_b128) and fed to v_wmma_f32_16x16x32_f16.
// Steady-state loop is branch-free: fill(next) / s_wait_asynccnt 3 / barrier /
// consume / barrier, with the last chunk peeled into an epilogue.
// Requires: M % 64 == 0, Npad % 128 == 0, K % 32 == 0, grid.x == (M/64)*(Npad/128).
// ---------------------------------------------------------------------------
#define A_HALVES (64 * 32)            // 2048 halves per A buffer
#define B_HALVES (128 * 32)           // 4096 halves per B buffer
#define BUF_HALVES (A_HALVES + B_HALVES)

template <bool HALF_OUT>
__global__ __launch_bounds__(256) void wmma_gemm_nt(
    const _Float16* __restrict__ A,   // [M, lda]
    const _Float16* __restrict__ BT,  // [Npad, ldb]  (B transposed)
    void* __restrict__ Cv,            // [M, ldc] float or _Float16
    int M, int Npad, int K, int lda, int ldb, int ldc, int Nlim) {
  __shared__ _Float16 smem[2 * BUF_HALVES];   // 24 KB, double buffered

  const int tid  = threadIdx.x;
  const int lane = tid & 31;
  const int wave = tid >> 5;
  const int mi   = wave & 3;        // 0..3  (16-row slice of the 64-row tile)
  const int ni   = wave >> 2;       // 0..1  (64-col slice of the 128-col tile)
  const int l15  = lane & 15;
  const int hi   = lane >> 4;

  const int nb      = Npad >> 7;                 // N tiles of 128
  const int blockM  = (blockIdx.x / nb) << 6;
  const int blockN  = (blockIdx.x % nb) << 7;

  // ---- async fill: each thread moves one 16B A chunk + two 16B B chunks ----
  const int ar = tid >> 2, ac = (tid & 3) * 8;           // A: row 0..63, k-chunk
  const _Float16* aSrc = A + (size_t)(blockM + ar) * lda + ac;
  const unsigned  aLds = (unsigned)(uintptr_t)&smem[ar * 32 + ac];
  const int br0 = tid >> 1;                              // B rows via 2 chunks
  const int bc0 = (tid & 1) * 16;                        // halves 0 / 16
  const _Float16* bSrc0 = BT + (size_t)(blockN + br0) * ldb + bc0;
  const unsigned  bLds0 = (unsigned)(uintptr_t)&smem[A_HALVES + br0 * 32 + bc0];
  const _Float16* bSrc1 = BT + (size_t)(blockN + 64 + br0) * ldb + bc0;
  const unsigned  bLds1 = (unsigned)(uintptr_t)&smem[A_HALVES + (64 + br0) * 32 + bc0];

  auto fill = [&](int kb, int buf) {
    unsigned long long ga = (unsigned long long)(uintptr_t)(aSrc + kb);
    unsigned la = aLds + (unsigned)(buf * BUF_HALVES * 2);
    asm volatile("global_load_async_to_lds_b128 %0, %1, off"
                 :: "v"(la), "v"(ga) : "memory");
    unsigned long long gb0 = (unsigned long long)(uintptr_t)(bSrc0 + kb);
    unsigned lb0 = bLds0 + (unsigned)(buf * BUF_HALVES * 2);
    asm volatile("global_load_async_to_lds_b128 %0, %1, off"
                 :: "v"(lb0), "v"(gb0) : "memory");
    unsigned long long gb1 = (unsigned long long)(uintptr_t)(bSrc1 + kb);
    unsigned lb1 = bLds1 + (unsigned)(buf * BUF_HALVES * 2);
    asm volatile("global_load_async_to_lds_b128 %0, %1, off"
                 :: "v"(lb1), "v"(gb1) : "memory");
  };

  v8f acc0 = {}, acc1 = {}, acc2 = {}, acc3 = {};
  const int arow = mi * 16 + l15;          // A fragment row within block tile
  const int bcol = ni * 64 + l15;          // first B fragment col within tile

  // Issue ALL fragment loads before the first WMMA so DScnt waits stagger.
  auto consume = [&](int buf) {
    const _Float16* As = smem + buf * BUF_HALVES;
    const _Float16* Bs = As + A_HALVES;
    union { v16h v; v8h h[2]; } a, b0, b1, b2, b3;
    a.h[0]  = *(const v8h*)&As[arow * 32 + hi * 16];
    a.h[1]  = *(const v8h*)&As[arow * 32 + hi * 16 + 8];
    b0.h[0] = *(const v8h*)&Bs[(bcol +  0) * 32 + hi * 16];
    b0.h[1] = *(const v8h*)&Bs[(bcol +  0) * 32 + hi * 16 + 8];
    b1.h[0] = *(const v8h*)&Bs[(bcol + 16) * 32 + hi * 16];
    b1.h[1] = *(const v8h*)&Bs[(bcol + 16) * 32 + hi * 16 + 8];
    b2.h[0] = *(const v8h*)&Bs[(bcol + 32) * 32 + hi * 16];
    b2.h[1] = *(const v8h*)&Bs[(bcol + 32) * 32 + hi * 16 + 8];
    b3.h[0] = *(const v8h*)&Bs[(bcol + 48) * 32 + hi * 16];
    b3.h[1] = *(const v8h*)&Bs[(bcol + 48) * 32 + hi * 16 + 8];
    acc0 = __builtin_amdgcn_wmma_f32_16x16x32_f16(false, a.v, false, b0.v,
                                                  (short)0, acc0, false, false);
    acc1 = __builtin_amdgcn_wmma_f32_16x16x32_f16(false, a.v, false, b1.v,
                                                  (short)0, acc1, false, false);
    acc2 = __builtin_amdgcn_wmma_f32_16x16x32_f16(false, a.v, false, b2.v,
                                                  (short)0, acc2, false, false);
    acc3 = __builtin_amdgcn_wmma_f32_16x16x32_f16(false, a.v, false, b3.v,
                                                  (short)0, acc3, false, false);
  };

  const int nChunks = K >> 5;
  fill(0, 0);
  for (int c = 0; c < nChunks - 1; ++c) {           // steady state: branch-free
    fill((c + 1) << 5, (c + 1) & 1);
    asm volatile("s_wait_asynccnt 0x3" ::: "memory");  // chunk c's fills done
    __syncthreads();
    consume(c & 1);
    __syncthreads();                                // consumed before refill
  }
  asm volatile("s_wait_asynccnt 0x0" ::: "memory"); // last chunk's fills done
  __syncthreads();
  consume((nChunks - 1) & 1);

  // ---- epilogue: C tile rows blockM+mi*16+{hi*8+r}, cols per sub-tile ----
  const int rbase = blockM + mi * 16 + hi * 8;
  v8f accs[4] = {acc0, acc1, acc2, acc3};
#pragma unroll
  for (int s = 0; s < 4; ++s) {
    const int col = blockN + ni * 64 + s * 16 + l15;
    if (col < Nlim) {
      if constexpr (HALF_OUT) {
        _Float16* C = (_Float16*)Cv;
#pragma unroll
        for (int r = 0; r < 8; ++r)
          C[(size_t)(rbase + r) * ldc + col] = (_Float16)accs[s][r];
      } else {
        float* C = (float*)Cv;
#pragma unroll
        for (int r = 0; r < 8; ++r)
          C[(size_t)(rbase + r) * ldc + col] = accs[s][r];
      }
    }
  }
}

// ---------------------------------------------------------------------------
// Casts / transposes
// ---------------------------------------------------------------------------
__global__ void cast_f32_to_f16(const float* __restrict__ s,
                                _Float16* __restrict__ d, long n) {
  long i = (long)blockIdx.x * blockDim.x + threadIdx.x;
  if (i < n) d[i] = (_Float16)s[i];
}

// dst[n,k] = src[k,n] (f32 [K,N] -> f16 [Npad,K], zero padded rows)
__global__ void transpose_cast_pad(const float* __restrict__ src,
                                   _Float16* __restrict__ dst,
                                   int K, int N, int Npad) {
  long i = (long)blockIdx.x * blockDim.x + threadIdx.x;
  long total = (long)Npad * K;
  if (i >= total) return;
  int n = (int)(i / K), k = (int)(i % K);
  dst[i] = (n < N) ? (_Float16)src[(size_t)k * N + n] : (_Float16)0.0f;
}

// ---------------------------------------------------------------------------
// Smolgen glue (small VALU kernels; ~11 GFLOP total)
// ---------------------------------------------------------------------------
__global__ void sg_special(const float* __restrict__ x, const float* __restrict__ w,
                           const float* __restrict__ b, float* __restrict__ out) {
  int bb = blockIdx.x, j = threadIdx.x;  // 16 threads
  const float* row = x + (size_t)bb * SL * EE;  // token 0
  float acc = b[j];
  for (int e = 0; e < EE; ++e) acc += row[e] * w[e * 16 + j];
  out[bb * 16 + j] = fmaxf(acc, 0.f);
}

__global__ void sg_conv0(const float* __restrict__ x, const float* __restrict__ w,
                         const float* __restrict__ b, float* __restrict__ h0) {
  int bb = blockIdx.x;
  const float* xb = x + (size_t)bb * SL * EE + EE;  // tokens 1..64 = [8][8][512]
  for (int o = threadIdx.x; o < 64 * 16; o += blockDim.x) {
    int pos = o >> 4, co = o & 15;
    int r = pos >> 3, c = pos & 7;
    float acc = b[co];
    for (int dr = -1; dr <= 1; ++dr) {
      int rr = r + dr; if (rr < 0 || rr > 7) continue;
      for (int dc = -1; dc <= 1; ++dc) {
        int cc = c + dc; if (cc < 0 || cc > 7) continue;
        const float* px = xb + (size_t)(rr * 8 + cc) * EE;
        const float* pw = w + ((size_t)((dr + 1) * 3 + (dc + 1)) * EE) * 16 + co;
        for (int e = 0; e < EE; ++e) acc += px[e] * pw[e * 16];
      }
    }
    h0[((size_t)bb * 64 + pos) * 16 + co] = fmaxf(acc, 0.f);
  }
}

__global__ void sg_conv1_mean(const float* __restrict__ h0, const float* __restrict__ w,
                              const float* __restrict__ b, float* __restrict__ bsum) {
  __shared__ float hs[64 * 32];
  int bb = blockIdx.x;
  const float* hb = h0 + (size_t)bb * 64 * 16;
  for (int o = threadIdx.x; o < 64 * 32; o += blockDim.x) {
    int pos = o >> 5, co = o & 31;
    int r = pos >> 3, c = pos & 7;
    float acc = b[co];
    for (int dr = -1; dr <= 1; ++dr) {
      int rr = r + dr; if (rr < 0 || rr > 7) continue;
      for (int dc = -1; dc <= 1; ++dc) {
        int cc = c + dc; if (cc < 0 || cc > 7) continue;
        const float* ph = hb + (rr * 8 + cc) * 16;
        const float* pw = w + ((dr + 1) * 3 + (dc + 1)) * 16 * 32 + co;
        for (int e = 0; e < 16; ++e) acc += ph[e] * pw[e * 32];
      }
    }
    hs[o] = fmaxf(acc, 0.f);
  }
  __syncthreads();
  if (threadIdx.x < 32) {
    float s = 0.f;
    for (int p = 0; p < 64; ++p) s += hs[p * 32 + threadIdx.x];
    bsum[bb * 32 + threadIdx.x] = s * (1.f / 64.f);
  }
}

__global__ void sg_g1(const float* __restrict__ x, const float* __restrict__ w,
                      const float* __restrict__ b, float* __restrict__ g1o) {
  int bb = blockIdx.x, j = threadIdx.x;  // 64 threads
  const float* g = x + (size_t)bb * SL * EE + 65 * EE;  // 14*512 flat
  float acc = b[j];
  for (int i = 0; i < 14 * EE; ++i) acc += g[i] * w[(size_t)i * 64 + j];
  g1o[bb * 64 + j] = fmaxf(acc, 0.f);
}

__global__ void sg_g2(const float* __restrict__ g1o, const float* __restrict__ w,
                      const float* __restrict__ b, float* __restrict__ g2o) {
  int bb = blockIdx.x, j = threadIdx.x;  // 32 threads
  float acc = b[j];
  for (int i = 0; i < 64; ++i) acc += g1o[bb * 64 + i] * w[i * 32 + j];
  g2o[bb * 32 + j] = fmaxf(acc, 0.f);
}

__global__ void sg_ps(const float* __restrict__ sp, const float* __restrict__ bs,
                      const float* __restrict__ g2o, const float* __restrict__ w,
                      const float* __restrict__ b, float* __restrict__ ps) {
  int bb = blockIdx.x, j = threadIdx.x;  // 128 threads
  float acc = b[j];
  for (int i = 0; i < 16; ++i) acc += sp[bb * 16 + i] * w[i * SUMD + j];
  for (int i = 0; i < 32; ++i) acc += bs[bb * 32 + i] * w[(16 + i) * SUMD + j];
  for (int i = 0; i < 32; ++i) acc += g2o[bb * 32 + i] * w[(48 + i) * SUMD + j];
  ps[bb * SUMD + j] = acc / (1.f + __expf(-acc));  // silu
}

__global__ void sg_hs(const float* __restrict__ ps, const float* __restrict__ hw,
                      const float* __restrict__ hb, _Float16* __restrict__ hsh) {
  int bh = blockIdx.x;           // B*H blocks
  int b = bh >> 3, h = bh & 7;
  int t = threadIdx.x;           // 128 threads
  float acc = hb[h * SUMD + t];
  const float* p = ps + (size_t)b * SUMD;
  const float* w = hw + (size_t)h * SUMD * SUMD;
  for (int s = 0; s < SUMD; ++s) acc += p[s] * w[s * SUMD + t];
  float v = acc / (1.f + __expf(-acc));
  hsh[(size_t)bh * SUMD + t] = (_Float16)v;
}

// ---------------------------------------------------------------------------
// Fused attention: per (b,h) block. LDS-staged Q/K/V (f16) + f32 score row.
// scores = QK^T/8 + smolgen logits; softmax; out = P @ V  (f16 output)
// ---------------------------------------------------------------------------
__global__ __launch_bounds__(128) void attn_softmax_pv(
    const _Float16* __restrict__ qh, const _Float16* __restrict__ kh,
    const _Float16* __restrict__ vh, const float* __restrict__ logits,
    _Float16* __restrict__ outh) {
  __shared__ _Float16 Qs[SL][DH];
  __shared__ _Float16 Ks[SL][DH];
  __shared__ _Float16 Vs[SL][DH];
  __shared__ float    Ps[SL][SL + 1];
  int bh = blockIdx.x;
  int b = bh >> 3, h = bh & 7;
  size_t base = (size_t)b * SL * (HH * DH) + h * DH;
  for (int i = threadIdx.x; i < SL * DH; i += blockDim.x) {
    int s = i >> 6, d = i & 63;
    size_t off = base + (size_t)s * (HH * DH) + d;
    Qs[s][d] = qh[off];
    Ks[s][d] = kh[off];
    Vs[s][d] = vh[off];
  }
  __syncthreads();
  int t = threadIdx.x;
  if (t < SL) {
    const float* lrow = logits + (size_t)bh * SS + t * SL;
    float mx = -1e30f;
    for (int T = 0; T < SL; ++T) {
      float s = 0.f;
      for (int d = 0; d < DH; ++d) s += (float)Qs[t][d] * (float)Ks[T][d];
      s = s * 0.125f + lrow[T];
      Ps[t][T] = s;
      mx = fmaxf(mx, s);
    }
    float sum = 0.f;
    for (int T = 0; T < SL; ++T) {
      float e = __expf(Ps[t][T] - mx);
      Ps[t][T] = e;
      sum += e;
    }
    float inv = 1.f / sum;
    size_t obase = (size_t)(b * SL + t) * (HH * DH) + h * DH;
    for (int d = 0; d < DH; ++d) {
      float acc = 0.f;
      for (int T = 0; T < SL; ++T) acc += Ps[t][T] * (float)Vs[T][d];
      outh[obase + d] = (_Float16)(acc * inv);
    }
  }
}

// ---------------------------------------------------------------------------
// Host orchestration
// ---------------------------------------------------------------------------
static inline char* bump(char*& p, size_t bytes) {
  char* r = p;
  p += (bytes + 255) & ~(size_t)255;
  return r;
}

extern "C" void kernel_launch(void* const* d_in, const int* in_sizes, int n_in,
                              void* d_out, int out_size, void* d_ws, size_t ws_size,
                              hipStream_t stream) {
  const float* x_q  = (const float*)d_in[0];
  const float* x_kv = (const float*)d_in[1];
  const float* wq   = (const float*)d_in[2];
  const float* wk   = (const float*)d_in[3];
  const float* wv   = (const float*)d_in[4];
  const float* wo   = (const float*)d_in[5];
  const float* sp_w = (const float*)d_in[6];
  const float* sp_b = (const float*)d_in[7];
  const float* c0w  = (const float*)d_in[8];
  const float* c0b  = (const float*)d_in[9];
  const float* c1w  = (const float*)d_in[10];
  const float* c1b  = (const float*)d_in[11];
  const float* g1w  = (const float*)d_in[12];
  const float* g1b  = (const float*)d_in[13];
  const float* g2w  = (const float*)d_in[14];
  const float* g2b  = (const float*)d_in[15];
  const float* sumw = (const float*)d_in[16];
  const float* sumb = (const float*)d_in[17];
  const float* hw   = (const float*)d_in[18];
  const float* hb   = (const float*)d_in[19];
  const float* spr  = (const float*)d_in[20];

  char* p = (char*)d_ws;
  const size_t NME = (size_t)MROWS * EE;  // 80896*512
  _Float16* Xq   = (_Float16*)bump(p, NME * 2);
  _Float16* Xkv  = (_Float16*)bump(p, NME * 2);
  _Float16* WqT  = (_Float16*)bump(p, (size_t)EE * EE * 2);
  _Float16* WkT  = (_Float16*)bump(p, (size_t)EE * EE * 2);
  _Float16* WvT  = (_Float16*)bump(p, (size_t)EE * EE * 2);
  _Float16* WoT  = (_Float16*)bump(p, (size_t)EE * EE * 2);
  _Float16* spT  = (_Float16*)bump(p, (size_t)SSPAD * SUMD * 2);
  _Float16* Qh   = (_Float16*)bump(p, NME * 2);
  _Float16* Kh   = (_Float16*)bump(p, NME * 2);
  _Float16* Vh   = (_Float16*)bump(p, NME * 2);
  _Float16* HSh  = (_Float16*)bump(p, (size_t)BB * HH * SUMD * 2);
  float*    LOG  = (float*)bump(p, (size_t)BB * HH * SS * 4);
  float*    SPC  = (float*)bump(p, (size_t)BB * 16 * 4);
  float*    H0   = (float*)bump(p, (size_t)BB * 64 * 16 * 4);
  float*    BSUM = (float*)bump(p, (size_t)BB * 32 * 4);
  float*    G1   = (float*)bump(p, (size_t)BB * 64 * 4);
  float*    G2   = (float*)bump(p, (size_t)BB * 32 * 4);
  float*    PS   = (float*)bump(p, (size_t)BB * SUMD * 4);
  _Float16* AO   = (_Float16*)bump(p, NME * 2);

  // 1) casts + weight transposes
  {
    long n = (long)NME;
    int blk = (int)((n + 255) / 256);
    cast_f32_to_f16<<<blk, 256, 0, stream>>>(x_q, Xq, n);
    cast_f32_to_f16<<<blk, 256, 0, stream>>>(x_kv, Xkv, n);
  }
  {
    long n = (long)EE * EE;
    int blk = (int)((n + 255) / 256);
    transpose_cast_pad<<<blk, 256, 0, stream>>>(wq, WqT, EE, EE, EE);
    transpose_cast_pad<<<blk, 256, 0, stream>>>(wk, WkT, EE, EE, EE);
    transpose_cast_pad<<<blk, 256, 0, stream>>>(wv, WvT, EE, EE, EE);
    transpose_cast_pad<<<blk, 256, 0, stream>>>(wo, WoT, EE, EE, EE);
  }
  {
    long n = (long)SSPAD * SUMD;
    int blk = (int)((n + 255) / 256);
    transpose_cast_pad<<<blk, 256, 0, stream>>>(spr, spT, SUMD, SS, SSPAD);
  }

  // 2) Q/K/V projections: [80896,512] x [512,512], f16 out
  {
    int blocks = (MROWS / 64) * (EE / 128);  // 1264 * 4 = 5056
    wmma_gemm_nt<true><<<blocks, 256, 0, stream>>>(Xq, WqT, Qh, MROWS, EE,
                                                   EE, EE, EE, EE, EE);
    wmma_gemm_nt<true><<<blocks, 256, 0, stream>>>(Xkv, WkT, Kh, MROWS, EE,
                                                   EE, EE, EE, EE, EE);
    wmma_gemm_nt<true><<<blocks, 256, 0, stream>>>(Xkv, WvT, Vh, MROWS, EE,
                                                   EE, EE, EE, EE, EE);
  }

  // 3) smolgen pipeline
  sg_special<<<BB, 16, 0, stream>>>(x_q, sp_w, sp_b, SPC);
  sg_conv0<<<BB, 256, 0, stream>>>(x_q, c0w, c0b, H0);
  sg_conv1_mean<<<BB, 256, 0, stream>>>(H0, c1w, c1b, BSUM);
  sg_g1<<<BB, 64, 0, stream>>>(x_q, g1w, g1b, G1);
  sg_g2<<<BB, 32, 0, stream>>>(G1, g2w, g2b, G2);
  sg_ps<<<BB, 128, 0, stream>>>(SPC, BSUM, G2, sumw, sumb, PS);
  sg_hs<<<BB * HH, 128, 0, stream>>>(PS, hw, hb, HSh);

  // 4) smolgen logits GEMM: [8192,128] x [128,6241] -> f32 [8192,6241]
  {
    int M = BB * HH;
    int blocks = (M / 64) * (SSPAD / 128);  // 128 * 49 = 6272
    wmma_gemm_nt<false><<<blocks, 256, 0, stream>>>(HSh, spT, LOG, M, SSPAD,
                                                    SUMD, SUMD, SUMD, SS, SS);
  }

  // 5) fused attention (scores + logits + softmax + PV), f16 out
  attn_softmax_pv<<<BB * HH, 128, 0, stream>>>(Qh, Kh, Vh, LOG, AO);

  // 6) output projection: [80896,512] x [512,512] -> f32 d_out
  {
    int blocks = (MROWS / 64) * (EE / 128);
    wmma_gemm_nt<false><<<blocks, 256, 0, stream>>>(AO, WoT, (float*)d_out,
                                                    MROWS, EE, EE, EE, EE,
                                                    EE, EE);
  }
}
